// BitNet_55679956025677
// MI455X (gfx1250) — 8-bit matrix operand order repaired and checked by reading, compiled, not run on hardware
//
#include <hip/hip_runtime.h>
#include <stdint.h>

#define BB   2
#define SS   2048
#define HID  2560
#define INTER 6912
#define NH   20
#define NKV  5
#define HD   128
#define MROWS (BB*SS)
#define EPSF 1e-5f

typedef _Float16 f16;
typedef unsigned int u32;
typedef __attribute__((ext_vector_type(4)))  u32      v4u;
typedef __attribute__((ext_vector_type(4)))  int      v4i;
typedef __attribute__((ext_vector_type(8)))  int      v8i;
typedef __attribute__((ext_vector_type(8)))  float    v8f;
typedef __attribute__((ext_vector_type(16))) _Float16 v16h;

union U16 { v16h v; f16 h[16]; float4 q[2]; };

#ifndef __has_builtin
#define __has_builtin(x) 0
#endif
#if __has_builtin(__builtin_amdgcn_tensor_load_to_lds) && __has_builtin(__builtin_amdgcn_s_wait_tensorcnt)
#define USE_TDM 1
#else
#define USE_TDM 0
#endif

#if USE_TDM
__device__ __forceinline__ void tdm_load_2d_i8(u32 lds_off, const void* gaddr,
                                               u32 tensor_d0, u32 tensor_d1,
                                               u32 stride_d0,
                                               u32 tile_d0, u32 tile_d1,
                                               u32 pad_interval, u32 pad_amount,
                                               u32 pad_en) {
    unsigned long long ga = (unsigned long long)gaddr;
    v4u g0;
    g0[0] = 1u;
    g0[1] = lds_off;
    g0[2] = (u32)ga;
    g0[3] = (u32)((ga >> 32) & 0x1FFFFFFu) | (2u << 30);
    v8i g1;
    g1[0] = (int)((pad_en ? (1u << 20) : 0u) |
                  (pad_interval << 22) | (pad_amount << 25));
    g1[1] = (int)((tensor_d0 & 0xFFFFu) << 16);
    g1[2] = (int)((tensor_d0 >> 16) | ((tensor_d1 & 0xFFFFu) << 16));
    g1[3] = (int)((tensor_d1 >> 16) | (tile_d0 << 16));
    g1[4] = (int)(tile_d1 & 0xFFFFu);
    g1[5] = (int)stride_d0;
    g1[6] = 0;
    g1[7] = 0;
    v4i z4 = {};
    v8i z8 = {};
    __builtin_amdgcn_tensor_load_to_lds(g0, g1, z4, z4, z8, 0);
}
#endif

__global__ void bn_init_acc(float* acc) {
    if (threadIdx.x < 8) acc[threadIdx.x] = 0.0f;
}

__global__ __launch_bounds__(256)
void bn_absmean(const float* __restrict__ W, size_t n, float* acc) {
    size_t stride = (size_t)gridDim.x * blockDim.x;
    float s = 0.0f;
    for (size_t i = (size_t)blockIdx.x * blockDim.x + threadIdx.x; i < n; i += stride)
        s += fabsf(W[i]);
    #pragma unroll
    for (int o = 16; o > 0; o >>= 1) s += __shfl_xor(s, o, 32);
    __shared__ float red[8];
    if ((threadIdx.x & 31) == 0) red[threadIdx.x >> 5] = s;
    __syncthreads();
    if (threadIdx.x == 0) {
        float t = 0.0f;
        #pragma unroll
        for (int i = 0; i < 8; i++) t += red[i];
        atomicAdd(acc, t);
    }
}

__global__ void bn_finalize_scale(const float* acc, float invn, float* out) {
    if (threadIdx.x == 0 && blockIdx.x == 0)
        *out = fmaxf(*acc * invn, 1e-5f);
}

__global__ __launch_bounds__(256)
void bn_quant_weight(const float* __restrict__ W, int8_t* __restrict__ Q,
                     size_t n, const float* wsc) {
    float s = 1.0f / (*wsc);
    size_t stride = (size_t)gridDim.x * blockDim.x;
    for (size_t i = (size_t)blockIdx.x * blockDim.x + threadIdx.x; i < n; i += stride) {
        float q = rintf(W[i] * s);
        q = fminf(fmaxf(q, -1.0f), 1.0f);
        Q[i] = (int8_t)q;
    }
}

__global__ __launch_bounds__(256)
void bn_rmsnorm_quant(const float* __restrict__ A, const float* __restrict__ Bm,
                      const float* __restrict__ gamma,
                      int8_t* __restrict__ Q, float* __restrict__ rs, int K) {
    __shared__ float sy[INTER];
    __shared__ float red[8];
    int row = blockIdx.x;
    const float* a = A + (size_t)row * K;
    const float* b = Bm ? Bm + (size_t)row * K : nullptr;
    int t = threadIdx.x;

    float ss = 0.0f;
    for (int i = t; i < K; i += 256) {
        float x = a[i];
        if (b) { float g = fmaxf(x, 0.0f); x = g * g * b[i]; }
        sy[i] = x;
        ss += x * x;
    }
    #pragma unroll
    for (int o = 16; o > 0; o >>= 1) ss += __shfl_xor(ss, o, 32);
    if ((t & 31) == 0) red[t >> 5] = ss;
    __syncthreads();
    if (t == 0) {
        float s = 0.0f;
        #pragma unroll
        for (int i = 0; i < 8; i++) s += red[i];
        red[0] = rsqrtf(s / (float)K + EPSF);
    }
    __syncthreads();
    float rinv = red[0];
    __syncthreads();

    float mx = 0.0f;
    for (int i = t; i < K; i += 256) {
        float y = gamma[i] * sy[i] * rinv;
        sy[i] = y;
        mx = fmaxf(mx, fabsf(y));
    }
    #pragma unroll
    for (int o = 16; o > 0; o >>= 1) mx = fmaxf(mx, __shfl_xor(mx, o, 32));
    if ((t & 31) == 0) red[t >> 5] = mx;
    __syncthreads();
    if (t == 0) {
        float m = 0.0f;
        #pragma unroll
        for (int i = 0; i < 8; i++) m = fmaxf(m, red[i]);
        m = fmaxf(m, 1e-5f);
        red[0] = m;
        rs[row] = m / 127.0f;
    }
    __syncthreads();
    float s = 127.0f / red[0];
    int8_t* q = Q + (size_t)row * K;
    for (int i = t; i < K; i += 256) {
        float v = rintf(sy[i] * s);
        v = fminf(fmaxf(v, -128.0f), 127.0f);
        q[i] = (int8_t)v;
    }
}

#define TLM 128
#define TLN 64
#define TLK 64
#define LSTR 80

__global__ __launch_bounds__(256)
void bn_gemm_iu8(const int8_t* __restrict__ A, const int8_t* __restrict__ W,
                 const float* __restrict__ rowScale, const float* __restrict__ wScale,
                 const float* __restrict__ residual, float* __restrict__ C,
                 int M, int N, int K) {
    __shared__ __align__(16) int8_t sA[TLM * LSTR];
    __shared__ __align__(16) int8_t sB[TLN * LSTR];

    int m0 = blockIdx.y * TLM, n0 = blockIdx.x * TLN;
    int t = threadIdx.x;
    int w = t >> 5;
    int wm = (w & 3) * 32, wn = (w >> 2) * 32;
    int lm = t & 15, hl = (t >> 4) & 1;

    v8i a00 = {}, a01 = {}, a10 = {}, a11 = {};

#if USE_TDM
    u32 ldsA = (u32)(uintptr_t)(void*)sA;
    u32 ldsB = (u32)(uintptr_t)(void*)sB;
    const int8_t* gA = A + (size_t)m0 * K;
    const int8_t* gB = W + (size_t)n0 * K;
#else
    int srA = t >> 1,  scA = (t & 1) * 32;
    int srB = t >> 2,  scB = (t & 3) * 16;
    const int4* gA = (const int4*)(A + (size_t)(m0 + srA) * K + scA);
    const int4* gB = (const int4*)(W + (size_t)(n0 + srB) * K + scB);
    int4* lA = (int4*)(sA + srA * LSTR + scA);
    int4* lB = (int4*)(sB + srB * LSTR + scB);
#endif

    for (int k0 = 0; k0 < K; k0 += TLK) {
#if USE_TDM
        if (w == 0) {
            tdm_load_2d_i8(ldsA, gA + k0, (u32)(K - k0), (u32)TLM, (u32)K,
                           TLK, TLM,  3,  3, 1);
            tdm_load_2d_i8(ldsB, gB + k0, (u32)(K - k0), (u32)TLN, (u32)K,
                           TLK, TLN, 3, 3, 1);
            __builtin_amdgcn_s_wait_tensorcnt((short)0);
        }
        __syncthreads();
#else
        lA[0] = gA[0];
        lA[1] = gA[1];
        lB[0] = gB[0];
        gA = (const int4*)((const int8_t*)gA + TLK);
        gB = (const int4*)((const int8_t*)gB + TLK);
        if (k0 + TLK < K) {
            __builtin_prefetch((const void*)gA, 0, 1);
            __builtin_prefetch((const void*)gB, 0, 1);
        }
        __syncthreads();
#endif
        v8i fa0, fa1, fb0, fb1;
        #pragma unroll
        for (int j = 0; j < 8; j++) {
            int ka = (j >> 1) * 16 + hl * 8 + (j & 1) * 4;
            fa0[j] = *(const int*)(sA + (wm +      lm) * LSTR + ka);
            fa1[j] = *(const int*)(sA + (wm + 16 + lm) * LSTR + ka);
            int kb = (j >> 1) * 16 + hl * 8 + (j & 1) * 4;
            fb0[j] = *(const int*)(sB + (wn +      lm) * LSTR + kb);
            fb1[j] = *(const int*)(sB + (wn + 16 + lm) * LSTR + kb);
        }
        a00 = __builtin_amdgcn_wmma_i32_16x16x64_iu8(true, fa0, true, fb0, a00, false, false);
        a01 = __builtin_amdgcn_wmma_i32_16x16x64_iu8(true, fa0, true, fb1, a01, false, false);
        a10 = __builtin_amdgcn_wmma_i32_16x16x64_iu8(true, fa1, true, fb0, a10, false, false);
        a11 = __builtin_amdgcn_wmma_i32_16x16x64_iu8(true, fa1, true, fb1, a11, false, false);
        __syncthreads();
    }

    float sw = *wScale;
    #pragma unroll
    for (int r = 0; r < 8; r++) {
        int row0 = m0 + wm + r + hl * 8;
        int row1 = row0 + 16;
        float s0 = rowScale[row0] * sw;
        float s1 = rowScale[row1] * sw;
        size_t i0 = (size_t)row0 * N + n0 + wn + lm;
        size_t i1 = (size_t)row1 * N + n0 + wn + lm;
        float v00 = (float)a00[r] * s0, v01 = (float)a01[r] * s0;
        float v10 = (float)a10[r] * s1, v11 = (float)a11[r] * s1;
        if (residual) {
            v00 += residual[i0]; v01 += residual[i0 + 16];
            v10 += residual[i1]; v11 += residual[i1 + 16];
        }
        C[i0] = v00; C[i0 + 16] = v01;
        C[i1] = v10; C[i1 + 16] = v11;
    }
}

__global__ __launch_bounds__(256)
void bn_rope_pack(const float* __restrict__ qf, const float* __restrict__ kf,
                  const float* __restrict__ vf, const float* __restrict__ cosb,
                  const float* __restrict__ sinb,
                  f16* __restrict__ qh, f16* __restrict__ kh, f16* __restrict__ vh) {
    int bs = blockIdx.x;
    int b = bs / SS, s = bs % SS;
    const float* cp = cosb + (size_t)bs * HD;
    const float* sp = sinb + (size_t)bs * HD;
    const float* qr = qf + (size_t)bs * (NH * HD);
    const float* kr = kf + (size_t)bs * (NKV * HD);
    const float* vr = vf + (size_t)bs * (NKV * HD);

    for (int i = threadIdx.x; i < NH * HD; i += blockDim.x) {
        int h = i >> 7, d = i & 127;
        float x  = qr[i];
        float ro = (d < 64) ? -qr[h * HD + d + 64] : qr[h * HD + d - 64];
        float y  = x * cp[d] + ro * sp[d];
        qh[(((size_t)b * NH + h) * SS + s) * HD + d] = (f16)y;
        if (i < NKV * HD) {
            float kx = kr[i];
            float ko = (d < 64) ? -kr[h * HD + d + 64] : kr[h * HD + d - 64];
            float ky = kx * cp[d] + ko * sp[d];
            size_t o = (((size_t)b * NKV + h) * SS + s) * HD + d;
            kh[o] = (f16)ky;
            vh[o] = (f16)vr[i];
        }
    }
}

__global__ __launch_bounds__(32)
void bn_attention(const f16* __restrict__ Qh, const f16* __restrict__ Kh,
                  const f16* __restrict__ Vh, float* __restrict__ Out) {
    int qt = blockIdx.x, h = blockIdx.y, b = blockIdx.z;
    int kvh = h / (NH / NKV);
    const f16* qp = Qh + (((size_t)b * NH  + h  ) * SS + qt * 16) * HD;
    const f16* kp = Kh + (((size_t)b * NKV + kvh) * SS) * HD;
    const f16* vp = Vh + (((size_t)b * NKV + kvh) * SS) * HD;

    int l = threadIdx.x, lm = l & 15, hl = l >> 4;

    __shared__ __align__(16) f16 sP[16 * 40];
    __shared__ __align__(16) f16 sV[32 * 136];

    U16 qa[4];
    #pragma unroll
    for (int c = 0; c < 4; c++) {
        const float4* p = (const float4*)(qp + (size_t)lm * HD + c * 32 + hl * 8);
        qa[c].q[0] = p[0];
        qa[c].q[1] = p[2];
    }

    float mrow[8], lrow[8];
    v8f o[8];
    #pragma unroll
    for (int r = 0; r < 8; r++) { mrow[r] = -1e30f; lrow[r] = 0.0f; }
    #pragma unroll
    for (int c = 0; c < 8; c++) o[c] = v8f{};

    const float sm = 0.08838834764831845f;

    for (int k0 = 0; k0 < SS; k0 += 32) {
        {
            const float4* src = (const float4*)(vp + (size_t)(k0 + l) * HD);
            float4* dst = (float4*)(sV + l * 136);
            #pragma unroll
            for (int i = 0; i < 16; i++) dst[i] = src[i];
        }

        v8f s0 = {}, s1 = {};
        #pragma unroll
        for (int c = 0; c < 4; c++) {
            U16 kb0, kb1;
            const float4* p0 = (const float4*)(kp + (size_t)(k0 + lm) * HD + c * 32 + hl * 8);
            kb0.q[0] = p0[0]; kb0.q[1] = p0[2];
            const float4* p1 = (const float4*)(kp + (size_t)(k0 + 16 + lm) * HD + c * 32 + hl * 8);
            kb1.q[0] = p1[0]; kb1.q[1] = p1[2];
            s0 = __builtin_amdgcn_wmma_f32_16x16x32_f16(false, qa[c].v, false, kb0.v, (short)0, s0, false, false);
            s1 = __builtin_amdgcn_wmma_f32_16x16x32_f16(false, qa[c].v, false, kb1.v, (short)0, s1, false, false);
        }

        float alpha[8];
        #pragma unroll
        for (int r = 0; r < 8; r++) {
            float a0 = s0[r] * sm, a1 = s1[r] * sm;
            float mx = fmaxf(a0, a1);
            #pragma unroll
            for (int o2 = 8; o2 > 0; o2 >>= 1) mx = fmaxf(mx, __shfl_xor(mx, o2, 16));
            float mnew = fmaxf(mrow[r], mx);
            alpha[r] = __expf(mrow[r] - mnew);
            mrow[r] = mnew;
            float p0 = __expf(a0 - mnew), p1 = __expf(a1 - mnew);
            float rsum = p0 + p1;
            #pragma unroll
            for (int o2 = 8; o2 > 0; o2 >>= 1) rsum += __shfl_xor(rsum, o2, 16);
            lrow[r] = lrow[r] * alpha[r] + rsum;
            sP[(r + hl * 8) * 40 + lm]      = (f16)p0;
            sP[(r + hl * 8) * 40 + 16 + lm] = (f16)p1;
        }

        #pragma unroll
        for (int c = 0; c < 8; c++)
            #pragma unroll
            for (int r = 0; r < 8; r++) o[c][r] *= alpha[r];

        U16 pa;
        {
            const float4* pp = (const float4*)(sP + lm * 40 + hl * 8);
            pa.q[0] = pp[0];
            pa.q[1] = pp[2];
        }
        #pragma unroll
        for (int c = 0; c < 8; c++) {
            U16 vb;
            #pragma unroll
            for (int e = 0; e < 16; e++) {
                int kk = (e >> 3) * 16 + hl * 8 + (e & 7);
                vb.h[e] = sV[kk * 136 + c * 16 + lm];
            }
            o[c] = __builtin_amdgcn_wmma_f32_16x16x32_f16(false, pa.v, false, vb.v, (short)0, o[c], false, false);
        }
    }

    float* op = Out + ((size_t)b * SS + qt * 16) * (NH * HD) + h * HD;
    #pragma unroll
    for (int r = 0; r < 8; r++) {
        float inv = 1.0f / lrow[r];
        int q = r + hl * 8;
        #pragma unroll
        for (int c = 0; c < 8; c++)
            op[(size_t)q * (NH * HD) + c * 16 + lm] = o[c][r] * inv;
    }
}

extern "C" void kernel_launch(void* const* d_in, const int* in_sizes, int n_in,
                              void* d_out, int out_size, void* d_ws, size_t ws_size,
                              hipStream_t stream) {
    const float* hidden  = (const float*)d_in[0];
    const float* cosb    = (const float*)d_in[2];
    const float* sinb    = (const float*)d_in[3];
    const float* w_in_ln = (const float*)d_in[4];
    const float* wq      = (const float*)d_in[5];
    const float* wk      = (const float*)d_in[6];
    const float* wv      = (const float*)d_in[7];
    const float* wo      = (const float*)d_in[8];
    const float* w_attn  = (const float*)d_in[9];
    const float* w_post  = (const float*)d_in[10];
    const float* w_gate  = (const float*)d_in[11];
    const float* w_up    = (const float*)d_in[12];
    const float* w_down  = (const float*)d_in[13];
    const float* w_ffn   = (const float*)d_in[14];
    float* out = (float*)d_out;

    char* p = (char*)d_ws;
    auto take = [&](size_t n) -> void* {
        void* r = (void*)p;
        p += (n + 255) & ~(size_t)255;
        return r;
    };

    float* acc    = (float*)take(8 * sizeof(float));
    float* wsc    = (float*)take(8 * sizeof(float));
    int8_t* wq8   = (int8_t*)take((size_t)NH * HD * HID);
    int8_t* wk8   = (int8_t*)take((size_t)NKV * HD * HID);
    int8_t* wv8   = (int8_t*)take((size_t)NKV * HD * HID);
    int8_t* wo8   = (int8_t*)take((size_t)HID * NH * HD);
    int8_t* wg8   = (int8_t*)take((size_t)INTER * HID);
    int8_t* wu8   = (int8_t*)take((size_t)INTER * HID);
    int8_t* wd8   = (int8_t*)take((size_t)HID * INTER);
    int8_t* xq    = (int8_t*)take((size_t)MROWS * HID);
    float*  xs    = (float*) take((size_t)MROWS * sizeof(float));
    int8_t* aq    = (int8_t*)take((size_t)MROWS * HID);
    float*  as_   = (float*) take((size_t)MROWS * sizeof(float));
    int8_t* x2q   = (int8_t*)take((size_t)MROWS * HID);
    float*  x2s   = (float*) take((size_t)MROWS * sizeof(float));
    int8_t* tq    = (int8_t*)take((size_t)MROWS * INTER);
    float*  ts    = (float*) take((size_t)MROWS * sizeof(float));
    float*  qf    = (float*) take((size_t)MROWS * NH * HD * sizeof(float));
    float*  kf    = (float*) take((size_t)MROWS * NKV * HD * sizeof(float));
    float*  vf    = (float*) take((size_t)MROWS * NKV * HD * sizeof(float));
    f16*    qh    = (f16*)   take((size_t)MROWS * NH * HD * sizeof(f16));
    f16*    kh    = (f16*)   take((size_t)MROWS * NKV * HD * sizeof(f16));
    f16*    vh    = (f16*)   take((size_t)MROWS * NKV * HD * sizeof(f16));
    float*  attn  = (float*) take((size_t)MROWS * NH * HD * sizeof(float));
    float*  h1    = (float*) take((size_t)MROWS * HID * sizeof(float));
    float*  G     = (float*) take((size_t)MROWS * INTER * sizeof(float));
    float*  U     = (float*) take((size_t)MROWS * INTER * sizeof(float));

    const float* wsrc[7] = { wq, wk, wv, wo, w_gate, w_up, w_down };
    int8_t* wdst[7] = { wq8, wk8, wv8, wo8, wg8, wu8, wd8 };
    size_t wn[7] = { (size_t)NH*HD*HID, (size_t)NKV*HD*HID, (size_t)NKV*HD*HID,
                     (size_t)HID*NH*HD, (size_t)INTER*HID, (size_t)INTER*HID,
                     (size_t)HID*INTER };

    bn_init_acc<<<1, 32, 0, stream>>>(acc);
    for (int i = 0; i < 7; i++)
        bn_absmean<<<256, 256, 0, stream>>>(wsrc[i], wn[i], acc + i);
    for (int i = 0; i < 7; i++)
        bn_finalize_scale<<<1, 32, 0, stream>>>(acc + i, 1.0f / (float)wn[i], wsc + i);
    for (int i = 0; i < 7; i++)
        bn_quant_weight<<<2048, 256, 0, stream>>>(wsrc[i], wdst[i], wn[i], wsc + i);

    bn_rmsnorm_quant<<<MROWS, 256, 0, stream>>>(hidden, nullptr, w_in_ln, xq, xs, HID);

    bn_gemm_iu8<<<dim3((NH*HD)/TLN,  MROWS/TLM), 256, 0, stream>>>(xq, wq8, xs, wsc+0, nullptr, qf, MROWS, NH*HD,  HID);
    bn_gemm_iu8<<<dim3((NKV*HD)/TLN, MROWS/TLM), 256, 0, stream>>>(xq, wk8, xs, wsc+1, nullptr, kf, MROWS, NKV*HD, HID);
    bn_gemm_iu8<<<dim3((NKV*HD)/TLN, MROWS/TLM), 256, 0, stream>>>(xq, wv8, xs, wsc+2, nullptr, vf, MROWS, NKV*HD, HID);

    bn_rope_pack<<<MROWS, 256, 0, stream>>>(qf, kf, vf, cosb, sinb, qh, kh, vh);

    bn_attention<<<dim3(SS/16, NH, BB), 32, 0, stream>>>(qh, kh, vh, attn);

    bn_rmsnorm_quant<<<MROWS, 256, 0, stream>>>(attn, nullptr, w_attn, aq, as_, NH*HD);
    bn_gemm_iu8<<<dim3(HID/TLN, MROWS/TLM), 256, 0, stream>>>(aq, wo8, as_, wsc+3, hidden, h1, MROWS, HID, NH*HD);

    bn_rmsnorm_quant<<<MROWS, 256, 0, stream>>>(h1, nullptr, w_post, x2q, x2s, HID);
    bn_gemm_iu8<<<dim3(INTER/TLN, MROWS/TLM), 256, 0, stream>>>(x2q, wg8, x2s, wsc+4, nullptr, G, MROWS, INTER, HID);
    bn_gemm_iu8<<<dim3(INTER/TLN, MROWS/TLM), 256, 0, stream>>>(x2q, wu8, x2s, wsc+5, nullptr, U, MROWS, INTER, HID);

    bn_rmsnorm_quant<<<MROWS, 256, 0, stream>>>(G, U, w_ffn, tq, ts, INTER);
    bn_gemm_iu8<<<dim3(HID/TLN, MROWS/TLM), 256, 0, stream>>>(tq, wd8, ts, wsc+6, h1, out, MROWS, HID, INTER);
}
